// CrossAttentionBlock_44289702756632
// MI455X (gfx1250) — compile-verified, matrix operand order checked by reading, not run on hardware
//
#include <hip/hip_runtime.h>
#include <hip/hip_bf16.h>
#include <math.h>

// CrossAttentionBlock for MI455X (gfx1250, wave32, WMMA).
//
// Key simplification (exact math): the single repeated key makes causal
// softmax uniform over the valid window, so attention output == V for every
// t.  Q / wq never influence the result.  The block reduces to:
//   vo[ba,:] = ((cx@wkv)[:,D:2D] + bkv[D:]) @ wo + bo          (tiny)
//   out      = x3 + gelu(x3@w_fc+b_fc)@w_pr + b_pr,  x3 = LN2(LN1(x)+vo)
// The 137-GFLOP MLP runs on v_wmma_f32_16x16x32_f16 with f32 accumulation;
// weights are pre-packed into fragment-ready f16 so each lane loads one
// contiguous 32B v16h per B-fragment.

typedef __attribute__((ext_vector_type(16))) _Float16 v16h;
typedef __attribute__((ext_vector_type(8)))  float    v8f;

#define Bd 8
#define Ad 128
#define Td 128
#define Dd 256
#define DFF 1024

// K-index inside a 32-wide k-step for half p (0..15) of lane-half hi (0/1),
// per CDNA5 ISA 16-bit A/B fragment layout (CDNA5 ISA, sec 7.12.2).
__device__ __forceinline__ int kOf(int p, int hi) {
    return (p < 8) ? (hi * 8 + p) : (16 + hi * 8 + (p - 8));
}
// Inverse: rem = K & 31 -> (hi, p)
__device__ __forceinline__ void invK(int rem, int& hi, int& p) {
    hi = (rem >> 3) & 1;
    p  = (rem & 7) + ((rem >> 4) << 3);
}

// ---------------------------------------------------------------------------
// Prologue A: pack w_fc (256x1024) and w_pr (1024x256) f32 -> fragment-ready
// f16.  Layout: frag[ntile][kstep][lane 0..31][p 0..15], value = W[K][N] with
// K = kstep*32 + kOf(p, lane>>4), N = ntile*16 + (lane&15).
// ---------------------------------------------------------------------------
__global__ void cab_pack_weights(const float* __restrict__ wfc,
                                 const float* __restrict__ wpr,
                                 _Float16* __restrict__ wfcP,
                                 _Float16* __restrict__ wprP) {
    int i = blockIdx.x * 256 + threadIdx.x;          // 0 .. 262143
    {   // w_fc: 64 ntiles x 8 ksteps x 32 lanes x 16 halves
        int p = i & 15, lane = (i >> 4) & 31, ks = (i >> 9) & 7, nt = i >> 12;
        int hi = lane >> 4;
        int K  = ks * 32 + kOf(p, hi);
        int N  = nt * 16 + (lane & 15);
        wfcP[i] = (_Float16)wfc[K * DFF + N];
    }
    {   // w_pr: 16 ntiles x 32 ksteps x 32 lanes x 16 halves
        int p = i & 15, lane = (i >> 4) & 31, ks = (i >> 9) & 31, nt = i >> 14;
        int hi = lane >> 4;
        int K  = ks * 32 + kOf(p, hi);
        int N  = nt * 16 + (lane & 15);
        wprP[i] = (_Float16)wpr[K * Dd + N];
    }
}

// ---------------------------------------------------------------------------
// Prologue B: vo[ba,d] = bo[d] + sum_k (bkv[D+k] + sum_j cx[ba,j]*wkv[j,D+k])
//                                * wo[k,d].   One block per ba (1024 blocks).
// ---------------------------------------------------------------------------
__global__ void cab_kv_vo(const float* __restrict__ cx,
                          const float* __restrict__ wkv,
                          const float* __restrict__ bkv,
                          const float* __restrict__ wo,
                          const float* __restrict__ bo,
                          float* __restrict__ vo) {
    __shared__ float Vs[Dd];
    int ba = blockIdx.x;
    int d  = threadIdx.x;                       // 0..255
    const float* cxr = cx + (size_t)ba * Dd;
    float acc = bkv[Dd + d];
    for (int k = 0; k < Dd; ++k)
        acc += cxr[k] * wkv[(size_t)k * (2 * Dd) + Dd + d];
    Vs[d] = acc;
    __syncthreads();
    float acc2 = bo[d];
    for (int k = 0; k < Dd; ++k)
        acc2 += Vs[k] * wo[(size_t)k * Dd + d];
    vo[(size_t)ba * Dd + d] = acc2;
}

// ---------------------------------------------------------------------------
// Main kernel: one workgroup (8 wave32) per 16-token tile.
//   Phase 1: LN1 + vo + LN2 -> x3 (f32 in LDS for residual, f16 packed A-frags)
//   Phase 2: H = gelu(x3 @ w_fc + b_fc) via WMMA, H packed back to LDS
//   Phase 3: out = x3 + H @ w_pr + b_pr via WMMA
// ---------------------------------------------------------------------------
__global__ void __launch_bounds__(256, 2)
cab_main(const float* __restrict__ x,
         const float* __restrict__ ln1w, const float* __restrict__ ln1b,
         const float* __restrict__ ln2w, const float* __restrict__ ln2b,
         const float* __restrict__ vo,
         const _Float16* __restrict__ wfcP,
         const _Float16* __restrict__ wprP,
         const float* __restrict__ bfc, const float* __restrict__ bpr,
         float* __restrict__ out) {
    __shared__ __align__(32) _Float16 aPack[8 * 32 * 16];    //  8 KB x3 frags
    __shared__ __align__(32) float    x3f[16 * Dd];          // 16 KB residual
    __shared__ __align__(32) _Float16 hPack[32 * 32 * 16];   // 32 KB H frags

    const int tid  = threadIdx.x;
    const int ba   = blockIdx.x >> 3;            // (b*A+a), 0..1023
    const int t0   = (blockIdx.x & 7) * 16;      // token tile base

    // ---------------- Phase 1: LayerNorms -------------------------------
    {
        const int m   = tid >> 4;                // token row 0..15
        const int sub = tid & 15;                // 16 lanes per token (in-wave)
        const float* xrow = x + (((size_t)ba * Td) + t0 + m) * Dd;
        const float* vor  = vo + (size_t)ba * Dd;

        float v[16];
        float s = 0.f;
        #pragma unroll
        for (int j = 0; j < 16; ++j) { v[j] = xrow[sub + 16 * j]; s += v[j]; }
        #pragma unroll
        for (int off = 8; off; off >>= 1) s += __shfl_xor(s, off, 16);
        float mu = s * (1.f / Dd);
        float ss = 0.f;
        #pragma unroll
        for (int j = 0; j < 16; ++j) { float t = v[j] - mu; ss += t * t; }
        #pragma unroll
        for (int off = 8; off; off >>= 1) ss += __shfl_xor(ss, off, 16);
        float rstd = rsqrtf(ss * (1.f / Dd) + 1e-5f);

        // x2 = LN1(x) + vo ; second pass statistics
        float s2 = 0.f;
        #pragma unroll
        for (int j = 0; j < 16; ++j) {
            int d = sub + 16 * j;
            float x1 = (v[j] - mu) * rstd * ln1w[d] + ln1b[d];
            v[j] = x1 + vor[d];
            s2 += v[j];
        }
        #pragma unroll
        for (int off = 8; off; off >>= 1) s2 += __shfl_xor(s2, off, 16);
        float mu2 = s2 * (1.f / Dd);
        float ss2 = 0.f;
        #pragma unroll
        for (int j = 0; j < 16; ++j) { float t = v[j] - mu2; ss2 += t * t; }
        #pragma unroll
        for (int off = 8; off; off >>= 1) ss2 += __shfl_xor(ss2, off, 16);
        float rstd2 = rsqrtf(ss2 * (1.f / Dd) + 1e-5f);

        #pragma unroll
        for (int j = 0; j < 16; ++j) {
            int d = sub + 16 * j;
            float x3 = (v[j] - mu2) * rstd2 * ln2w[d] + ln2b[d];
            x3f[m * Dd + d] = x3;
            int ks = d >> 5, rem = d & 31, hi, p;
            invK(rem, hi, p);
            aPack[(ks * 32 + (hi * 16 + m)) * 16 + p] = (_Float16)x3;
        }
    }
    __syncthreads();

    // ---------------- Phase 2: FC GEMM 16x1024, K=256 + GeLU ------------
    const int wave = tid >> 5, lane = tid & 31;
    const int nlo = lane & 15, hi = lane >> 4;
    {
        v16h aF[8];
        #pragma unroll
        for (int ks = 0; ks < 8; ++ks)
            aF[ks] = *(const v16h*)&aPack[(ks * 32 + lane) * 16];

        #pragma unroll
        for (int it = 0; it < 8; ++it) {
            int nt = wave * 8 + it;                        // N-tile 0..63
            const _Float16* bptr = wfcP + ((size_t)nt * 8 * 32 + lane) * 16;
            __builtin_prefetch(bptr + 8 * 32 * 16, 0, 1); // next tile's frags
            v8f acc = {};
            #pragma unroll
            for (int ks = 0; ks < 8; ++ks) {
                v16h bF = *(const v16h*)(bptr + (size_t)ks * 32 * 16);
                acc = __builtin_amdgcn_wmma_f32_16x16x32_f16(
                    false, aF[ks], false, bF, (short)0, acc, false, false);
            }
            int col  = nt * 16 + nlo;                     // 0..1023
            float bb = bfc[col];
            int ks2 = col >> 5, hi2, p2;
            invK(col & 31, hi2, p2);
            #pragma unroll
            for (int r = 0; r < 8; ++r) {
                int mr  = r + 8 * hi;                     // token row
                float h = acc[r] + bb;
                float g = 0.5f * h * (1.f + erff(h * 0.70710678f)); // exact gelu
                hPack[(ks2 * 32 + (hi2 * 16 + mr)) * 16 + p2] = (_Float16)g;
            }
        }
    }
    __syncthreads();

    // ---------------- Phase 3: PR GEMM 16x256, K=1024 + residual --------
    {
        #pragma unroll
        for (int it = 0; it < 2; ++it) {
            int nt = wave * 2 + it;                       // N-tile 0..15
            const _Float16* bptr = wprP + ((size_t)nt * 32 * 32 + lane) * 16;
            v8f acc = {};
            #pragma unroll
            for (int ks = 0; ks < 32; ++ks) {
                v16h aF = *(const v16h*)&hPack[(ks * 32 + lane) * 16];
                v16h bF = *(const v16h*)(bptr + (size_t)ks * 32 * 16);
                acc = __builtin_amdgcn_wmma_f32_16x16x32_f16(
                    false, aF, false, bF, (short)0, acc, false, false);
            }
            int n2   = nt * 16 + nlo;                     // output column
            float bb = bpr[n2];
            #pragma unroll
            for (int r = 0; r < 8; ++r) {
                int mr  = r + 8 * hi;
                float o = acc[r] + bb + x3f[mr * Dd + n2];
                out[(((size_t)ba * Td) + t0 + mr) * Dd + n2] = o;
            }
        }
    }
}

// ---------------------------------------------------------------------------
extern "C" void kernel_launch(void* const* d_in, const int* in_sizes, int n_in,
                              void* d_out, int out_size, void* d_ws, size_t ws_size,
                              hipStream_t stream) {
    (void)in_sizes; (void)n_in; (void)out_size; (void)ws_size;
    const float* x    = (const float*)d_in[0];
    const float* cx   = (const float*)d_in[1];
    const float* ln1w = (const float*)d_in[2];
    const float* ln1b = (const float*)d_in[3];
    // d_in[4] = wq, d_in[5] = bq : provably unused (attention output == V)
    const float* wkv  = (const float*)d_in[6];
    const float* bkv  = (const float*)d_in[7];
    const float* wo   = (const float*)d_in[8];
    const float* bo   = (const float*)d_in[9];
    const float* ln2w = (const float*)d_in[10];
    const float* ln2b = (const float*)d_in[11];
    const float* wfc  = (const float*)d_in[12];
    const float* bfc  = (const float*)d_in[13];
    const float* wpr  = (const float*)d_in[14];
    const float* bpr  = (const float*)d_in[15];
    float* out = (float*)d_out;

    // workspace layout (needs 2 MB): packed f16 weights + vo
    char* ws = (char*)d_ws;
    _Float16* wfcP = (_Float16*)(ws);                       // 512 KB
    _Float16* wprP = (_Float16*)(ws + 512 * 1024);          // 512 KB
    float*    vo   = (float*)   (ws + 1024 * 1024);         //   1 MB

    cab_pack_weights<<<1024, 256, 0, stream>>>(wfc, wpr, wfcP, wprP);
    cab_kv_vo<<<Bd * Ad, Dd, 0, stream>>>(cx, wkv, bkv, wo, bo, vo);
    cab_main<<<Bd * Ad * (Td / 16), 256, 0, stream>>>(
        x, ln1w, ln1b, ln2w, ln2b, vo, wfcP, wprP, bfc, bpr, out);
}
